// Attention_32658931319237
// MI455X (gfx1250) — matrix operand order repaired and checked by reading, not compiled, not run on hardware
//
#include <hip/hip_runtime.h>
#include <hip/hip_bf16.h>
#include <math.h>

// ---------------------------------------------------------------------------
// Types for CDNA5 WMMA bf16 path
// ---------------------------------------------------------------------------
typedef __attribute__((ext_vector_type(16))) __bf16 bf16x16;
typedef __attribute__((ext_vector_type(8)))  __bf16 bf16x8;
typedef __attribute__((ext_vector_type(8)))  float  f32x8;

union Frag16 { bf16x16 v; bf16x8 h[2]; };

__device__ __forceinline__ f32x8 wmma_bf16(bf16x16 a, bf16x16 b, f32x8 c) {
  // D = A(16x32) * B(32x16) + C(16x16), fp32 accumulate
  return __builtin_amdgcn_wmma_f32_16x16x32_bf16(
      /*neg_a=*/false, a, /*neg_b=*/false, b,
      /*c_mod=*/(short)0, c, /*reuse_a=*/false, /*reuse_b=*/false);
}

__device__ __forceinline__ f32x8 zero_f32x8() {
  f32x8 r;
#pragma unroll
  for (int i = 0; i < 8; ++i) r[i] = 0.0f;
  return r;
}
__device__ __forceinline__ bf16x16 zero_bf16x16() {
  bf16x16 r;
#pragma unroll
  for (int i = 0; i < 16; ++i) r[i] = (__bf16)0.0f;
  return r;
}

// ---------------------------------------------------------------------------
// Problem dimensions (fixed by the reference)
// ---------------------------------------------------------------------------
#define BATCH 8
#define CDIM 128
#define HW 16384            // 128*128
#define NHEADS 4
#define HCH 32              // CDIM / NHEADS

// Workspace layout (bytes), all bf16 tensors unless noted
#define OFF_XT     ((size_t)0)            // x transposed  [b][pos][c]      33.5MB
#define OFF_XL     ((size_t)33554432)     // conv out      [b][pos][c]      33.5MB
#define OFF_Q      ((size_t)67108864)     // q             [b][h][ch][pos]  33.5MB
#define OFF_K      ((size_t)100663296)    // k             [b][h][ch][pos]  33.5MB
#define OFF_V      ((size_t)134217728)    // v             [b][h][pos][ch]  33.5MB
#define OFF_ATTN   ((size_t)167772160)    // attn          [b][h][32][32]   64KB
#define OFF_INVN   ((size_t)167837696)    // f32 inv-norms [b][256]         8KB
#define OFF_WCONV  ((size_t)167845888)    // conv A-frags  [9][8][4][32][16]
#define OFF_WQKV   ((size_t)168140800)    // qkv  A-frags  [24][4][32][16]
#define OFF_WPROJ  ((size_t)168239104)    // proj A-frags  [8][4][32][16]

#define N_CONV_FRAG (9*8*4*32*16)    // 147456
#define N_QKV_FRAG  (24*4*32*16)     //  49152
#define N_PROJ_FRAG (8*4*32*16)      //  16384

// ---------------------------------------------------------------------------
// K0: pre-pack weights into WMMA A-fragment lane order (bf16).
// A-frag mapping (16x16x32 bf16): frag[lane][e] = W[m0+lane%16][k0+(lane/16)*16+e]
// ---------------------------------------------------------------------------
__global__ void prep_weights(const float* __restrict__ wl,   // [128][128][3][3]
                             const float* __restrict__ wq,   // [384][128]
                             const float* __restrict__ wp,   // [128][128]
                             __bf16* __restrict__ cf,
                             __bf16* __restrict__ qf,
                             __bf16* __restrict__ pf) {
  int t = blockIdx.x * 256 + threadIdx.x;
  if (t < N_CONV_FRAG) {
    int e = t & 15, lane = (t >> 4) & 31, kc = (t >> 9) & 3, mt = (t >> 11) & 7, tap = t >> 14;
    int m = mt * 16 + (lane & 15);
    int k = kc * 32 + (lane >> 4) * 16 + e;
    cf[t] = (__bf16)wl[((m * 128 + k) * 3 + tap / 3) * 3 + (tap % 3)];
  } else if (t < N_CONV_FRAG + N_QKV_FRAG) {
    int t2 = t - N_CONV_FRAG;
    int e = t2 & 15, lane = (t2 >> 4) & 31, kc = (t2 >> 9) & 3, mt = t2 >> 11;  // mt < 24
    int m = mt * 16 + (lane & 15);
    int k = kc * 32 + (lane >> 4) * 16 + e;
    qf[t2] = (__bf16)wq[m * 128 + k];
  } else if (t < N_CONV_FRAG + N_QKV_FRAG + N_PROJ_FRAG) {
    int t3 = t - N_CONV_FRAG - N_QKV_FRAG;
    int e = t3 & 15, lane = (t3 >> 4) & 31, kc = (t3 >> 9) & 3, mt = t3 >> 11;
    int m = mt * 16 + (lane & 15);
    int k = kc * 32 + (lane >> 4) * 16 + e;
    pf[t3] = (__bf16)wp[m * 128 + k];
  }
}

// ---------------------------------------------------------------------------
// K1: x (NCHW f32) -> XT [b][pos][c] bf16 (channel-contiguous for B-frags)
// ---------------------------------------------------------------------------
__global__ void xpose(const float* __restrict__ x, __bf16* __restrict__ xt) {
  size_t t = (size_t)blockIdx.x * 256 + threadIdx.x;  // 8*128*16384 threads
  int n = (int)(t & 16383);
  int c = (int)((t >> 14) & 127);
  int b = (int)(t >> 21);
  xt[((size_t)b * HW + n) * CDIM + c] = (__bf16)x[t];  // coalesced read
}

// ---------------------------------------------------------------------------
// K2: 3x3 SAME conv as implicit GEMM, 9 taps x 4 K-chunks of WMMA.
// grid (128 nblocks, 8 mtiles, 8 batch), block 256 (8 waves, 16 positions each)
// ---------------------------------------------------------------------------
__global__ void conv3x3(const __bf16* __restrict__ xt, const __bf16* __restrict__ cf,
                        __bf16* __restrict__ xl) {
  int lane = threadIdx.x & 31, wave = threadIdx.x >> 5;
  int half = lane >> 4;
  int b = blockIdx.z, mt = blockIdx.y;
  int nn = blockIdx.x * 128 + wave * 16 + (lane & 15);
  int y = nn >> 7, x = nn & 127;

  f32x8 acc = zero_f32x8();
#pragma unroll
  for (int tap = 0; tap < 9; ++tap) {
    int yy = y + tap / 3 - 1, xx = x + tap % 3 - 1;
    bool inb = ((unsigned)yy < 128u) && ((unsigned)xx < 128u);
    int sp = inb ? (yy * 128 + xx) : 0;
    const __bf16* bp = xt + ((size_t)b * HW + sp) * CDIM + half * 16;
#pragma unroll
    for (int kc = 0; kc < 4; ++kc) {
      bf16x16 a = *(const bf16x16*)(cf + (size_t)(((tap * 8 + mt) * 4 + kc) * 32 + lane) * 16);
      bf16x16 bb = inb ? *(const bf16x16*)(bp + kc * 32) : zero_bf16x16();
      acc = wmma_bf16(a, bb, acc);
    }
  }
  bf16x8 o;
#pragma unroll
  for (int i = 0; i < 8; ++i) o[i] = (__bf16)acc[i];
  *(bf16x8*)(xl + ((size_t)b * HW + nn) * CDIM + mt * 16 + half * 8) = o;
}

// ---------------------------------------------------------------------------
// K3: qkv = w_qkv(384x128) @ x_local; writes q/k channel-major, v pos-major.
// grid (128 nblocks, 24 mtiles, 8 batch), block 256
// ---------------------------------------------------------------------------
__global__ void qkv_gemm(const __bf16* __restrict__ xl, const __bf16* __restrict__ qfr,
                         __bf16* __restrict__ Q, __bf16* __restrict__ K,
                         __bf16* __restrict__ V) {
  int lane = threadIdx.x & 31, wave = threadIdx.x >> 5;
  int half = lane >> 4;
  int b = blockIdx.z, mt = blockIdx.y;
  int nn = blockIdx.x * 128 + wave * 16 + (lane & 15);

  const __bf16* bp = xl + ((size_t)b * HW + nn) * CDIM + half * 16;
  f32x8 acc = zero_f32x8();
#pragma unroll
  for (int kc = 0; kc < 4; ++kc) {
    bf16x16 a = *(const bf16x16*)(qfr + (size_t)((mt * 4 + kc) * 32 + lane) * 16);
    bf16x16 bb = *(const bf16x16*)(bp + kc * 32);
    acc = wmma_bf16(a, bb, acc);
  }

  int m0 = mt * 16 + half * 8;  // global output channel of acc[0]
  if (mt < 8) {                 // q -> [b][h][ch][pos]
    int head = m0 >> 5, ch = m0 & 31;
    __bf16* qp = Q + ((size_t)((b * NHEADS + head) * HCH + ch)) * HW + nn;
#pragma unroll
    for (int i = 0; i < 8; ++i) qp[(size_t)i * HW] = (__bf16)acc[i];
  } else if (mt < 16) {         // k -> [b][h][ch][pos]
    int g = m0 - 128;
    int head = g >> 5, ch = g & 31;
    __bf16* kp = K + ((size_t)((b * NHEADS + head) * HCH + ch)) * HW + nn;
#pragma unroll
    for (int i = 0; i < 8; ++i) kp[(size_t)i * HW] = (__bf16)acc[i];
  } else {                      // v -> [b][h][pos][ch]
    int g = m0 - 256;
    int head = g >> 5, ch = g & 31;
    bf16x8 o;
#pragma unroll
    for (int i = 0; i < 8; ++i) o[i] = (__bf16)acc[i];
    *(bf16x8*)(V + ((size_t)((b * NHEADS + head) * HW + nn)) * HCH + ch) = o;
  }
}

// ---------------------------------------------------------------------------
// K4: inverse L2 norms of q/k rows (spatial axis). 2048 rows, 256 thr/block.
// ---------------------------------------------------------------------------
__global__ void inv_norms(const __bf16* __restrict__ Q, const __bf16* __restrict__ K,
                          float* __restrict__ invn) {
  __shared__ float red[256];
  int r = blockIdx.x;        // [b][256] ; 0-127 = q channels, 128-255 = k channels
  int b = r >> 8, idx = r & 255;
  const __bf16* p = (idx < 128) ? (Q + ((size_t)b * 128 + idx) * HW)
                                : (K + ((size_t)b * 128 + (idx - 128)) * HW);
  float s = 0.0f;
  for (int j = threadIdx.x; j < HW; j += 256) {
    float v = (float)p[j];
    s += v * v;
  }
  red[threadIdx.x] = s;
  __syncthreads();
  for (int off = 128; off > 0; off >>= 1) {
    if (threadIdx.x < off) red[threadIdx.x] += red[threadIdx.x + off];
    __syncthreads();
  }
  if (threadIdx.x == 0)
    invn[r] = 1.0f / fmaxf(sqrtf(red[0]), 1e-12f);
}

// ---------------------------------------------------------------------------
// K5: attn = softmax( (q @ k^T) * invq * invk^T ) per (b, head).
// grid 32, block 128 (4 waves; wave w -> 16x16 tile (w>>1, w&1)); K = 16384.
// ---------------------------------------------------------------------------
__global__ void attn_kernel(const __bf16* __restrict__ Q, const __bf16* __restrict__ K,
                            const float* __restrict__ invn, __bf16* __restrict__ ATTN) {
  __shared__ float lg[32][32];
  int lane = threadIdx.x & 31, wave = threadIdx.x >> 5;
  int half = lane >> 4;
  int b = blockIdx.x >> 2, h = blockIdx.x & 3;
  int m0 = (wave >> 1) * 16, n0 = (wave & 1) * 16;
  int row = m0 + (lane & 15);  // q channel within head
  int col = n0 + (lane & 15);  // k channel within head

  const __bf16* qp = Q + (size_t)((b * NHEADS + h) * HCH) * HW;
  const __bf16* kp = K + (size_t)((b * NHEADS + h) * HCH) * HW;

  f32x8 acc = zero_f32x8();
  for (int k0 = 0; k0 < HW; k0 += 32) {
    Frag16 a;
    a.h[0] = *(const bf16x8*)(qp + (size_t)row * HW + k0 + half * 16);
    a.h[1] = *(const bf16x8*)(qp + (size_t)row * HW + k0 + half * 16 + 8);
    bf16x16 bb = *(const bf16x16*)(kp + (size_t)col * HW + k0 + half * 16);
    acc = wmma_bf16(a.v, bb, acc);
  }

  float invk = invn[b * 256 + 128 + h * HCH + col];
#pragma unroll
  for (int i = 0; i < 8; ++i) {
    int m = m0 + half * 8 + i;
    float invq = invn[b * 256 + h * HCH + m];
    lg[m][col] = acc[i] * invq * invk;
  }
  __syncthreads();

  if (threadIdx.x < 32) {
    int rr = threadIdx.x;
    float mx = -INFINITY;
    for (int j = 0; j < 32; ++j) mx = fmaxf(mx, lg[rr][j]);
    float e[32], s = 0.0f;
    for (int j = 0; j < 32; ++j) { e[j] = __expf(lg[rr][j] - mx); s += e[j]; }
    float inv = 1.0f / s;
    __bf16* ap = ATTN + (size_t)(((b * NHEADS + h) * HCH + rr)) * HCH;
    for (int j = 0; j < 32; ++j) ap[j] = (__bf16)(e[j] * inv);
  }
}

// ---------------------------------------------------------------------------
// K6: out = attn @ v (per head, K=32), then final = w_proj @ out (K=128),
// fused through LDS. grid (1024 nblocks of 16 pos, 8 batch), block 256.
// ---------------------------------------------------------------------------
__global__ void av_proj(const __bf16* __restrict__ ATTN, const __bf16* __restrict__ V,
                        const __bf16* __restrict__ pfr, float* __restrict__ out) {
  __shared__ __align__(32) __bf16 ol[16][CDIM];  // [n_local][c] bf16
  int lane = threadIdx.x & 31, wave = threadIdx.x >> 5;
  int half = lane >> 4;
  int b = blockIdx.y;
  int n0 = blockIdx.x * 16;
  int nl = lane & 15;

  // Stage 1: wave w computes out channels [w*16, w*16+16) for these 16 positions
  {
    int c0 = wave * 16;
    int head = c0 >> 5;
    int r = (wave & 1) * 16 + nl;  // attn row within head
    const __bf16* ap = ATTN + (size_t)((b * NHEADS + head) * HCH) * HCH;
    Frag16 a;
    a.h[0] = *(const bf16x8*)(ap + r * HCH + half * 16);
    a.h[1] = *(const bf16x8*)(ap + r * HCH + half * 16 + 8);
    bf16x16 bb = *(const bf16x16*)(V + ((size_t)((b * NHEADS + head) * HW + n0 + nl)) * HCH + half * 16);
    f32x8 acc = wmma_bf16(a.v, bb, zero_f32x8());
    bf16x8 o;
#pragma unroll
    for (int i = 0; i < 8; ++i) o[i] = (__bf16)acc[i];
    *(bf16x8*)(&ol[nl][c0 + half * 8]) = o;
  }
  __syncthreads();

  // Stage 2: final = w_proj @ out ; wave w computes output channels [w*16, +16)
  {
    int o0 = wave * 16;
    f32x8 acc = zero_f32x8();
#pragma unroll
    for (int kc = 0; kc < 4; ++kc) {
      bf16x16 a = *(const bf16x16*)(pfr + (size_t)((wave * 4 + kc) * 32 + lane) * 16);
      bf16x16 bb = *(const bf16x16*)(&ol[nl][kc * 32 + half * 16]);
      acc = wmma_bf16(a, bb, acc);
    }
#pragma unroll
    for (int i = 0; i < 8; ++i) {
      int oc = o0 + half * 8 + i;
      out[((size_t)b * CDIM + oc) * HW + n0 + nl] = acc[i];
    }
  }
}

// ---------------------------------------------------------------------------
// Launch
// ---------------------------------------------------------------------------
extern "C" void kernel_launch(void* const* d_in, const int* in_sizes, int n_in,
                              void* d_out, int out_size, void* d_ws, size_t ws_size,
                              hipStream_t stream) {
  (void)in_sizes; (void)n_in; (void)out_size; (void)ws_size;
  const float* x  = (const float*)d_in[0];  // [8][128][128][128]
  const float* wl = (const float*)d_in[1];  // [128][128][3][3]
  const float* wq = (const float*)d_in[2];  // [384][128]
  const float* wp = (const float*)d_in[3];  // [128][128]
  float* outp = (float*)d_out;

  char* ws = (char*)d_ws;
  __bf16* XT   = (__bf16*)(ws + OFF_XT);
  __bf16* XL   = (__bf16*)(ws + OFF_XL);
  __bf16* Q    = (__bf16*)(ws + OFF_Q);
  __bf16* K    = (__bf16*)(ws + OFF_K);
  __bf16* V    = (__bf16*)(ws + OFF_V);
  __bf16* ATN  = (__bf16*)(ws + OFF_ATTN);
  float*  INVN = (float*) (ws + OFF_INVN);
  __bf16* WC   = (__bf16*)(ws + OFF_WCONV);
  __bf16* WQ   = (__bf16*)(ws + OFF_WQKV);
  __bf16* WP   = (__bf16*)(ws + OFF_WPROJ);

  prep_weights<<<(N_CONV_FRAG + N_QKV_FRAG + N_PROJ_FRAG) / 256, 256, 0, stream>>>(
      wl, wq, wp, WC, WQ, WP);
  xpose<<<(BATCH * CDIM * HW) / 256, 256, 0, stream>>>(x, XT);
  conv3x3<<<dim3(128, 8, BATCH), 256, 0, stream>>>(XT, WC, XL);
  qkv_gemm<<<dim3(128, 24, BATCH), 256, 0, stream>>>(XL, WQ, Q, K, V);
  inv_norms<<<BATCH * 256, 256, 0, stream>>>(Q, K, INVN);
  attn_kernel<<<BATCH * NHEADS, 128, 0, stream>>>(Q, K, INVN, ATN);
  av_proj<<<dim3(HW / 16, BATCH), 256, 0, stream>>>(ATN, V, WP, outp);
}
